// SAGEModelBen1_27152783245337
// MI455X (gfx1250) — matrix operand order repaired and checked by reading, not compiled, not run on hardware
//
#include <hip/hip_runtime.h>
#include <hip/hip_bf16.h>

typedef __attribute__((ext_vector_type(16))) _Float16 v16h;
typedef __attribute__((ext_vector_type(8)))  float    v8f;

#define N_NODES 50000
#define N_EDGES 800000
#define IN_DIM   64
#define HID_DIM 128
#define OUT_DIM  40
#define K1DIM   128   // layer1 concat: [mean(64) | x(64)]
#define K2DIM   256   // layer2 concat: [mean2(128) | h1(128)]
#define LDS1_STRIDE (K1DIM + 8)   // +16B pad: rotate 4 banks/row -> conflict-free b128
#define LDS2_STRIDE (K2DIM + 8)

union AFrag { v16h v; uint4 q[2]; };

// ---------------------------------------------------------------------------
// Weight pre-swizzle into the exact dense-B (32x16 f16) per-lane WMMA layout:
// buffer index = ((ot*NIT + it)*32 + lane)*16 + e
// element value = Wc[o = ot*16 + (lane&15)][k = it*32 + (lane<16?0:8) + (e<8 ? e : 16 + e - 8)]
// ---------------------------------------------------------------------------
__global__ __launch_bounds__(256) void prep_w1_kernel(
    const float* __restrict__ W_l1, const float* __restrict__ W_r1,
    _Float16* __restrict__ w1sw)
{
    int gid = blockIdx.x * 256 + threadIdx.x;          // 16384 = 8ot*4it*32*16
    int e    = gid & 15;
    int lane = (gid >> 4) & 31;
    int it   = (gid >> 9) & 3;
    int ot   = gid >> 11;
    int k = it * 32 + ((lane < 16) ? 0 : 8) + ((e < 8) ? e : 16 + (e - 8));      // 0..127
    int o = ot * 16 + (lane & 15);                     // 0..127
    float v = (k < IN_DIM) ? W_l1[o * IN_DIM + k] : W_r1[o * IN_DIM + (k - IN_DIM)];
    w1sw[gid] = (_Float16)v;
}

__global__ __launch_bounds__(256) void prep_w2_kernel(
    const float* __restrict__ W_l2, const float* __restrict__ W_r2,
    _Float16* __restrict__ w2sw)
{
    int gid = blockIdx.x * 256 + threadIdx.x;          // 12288 = 3ot*8it*32*16
    int e    = gid & 15;
    int lane = (gid >> 4) & 31;
    int it   = (gid >> 9) & 7;
    int ot   = gid >> 12;
    int k = it * 32 + ((lane < 16) ? 0 : 8) + ((e < 8) ? e : 16 + (e - 8));      // 0..255
    int o = ot * 16 + (lane & 15);                     // 0..47 (40..47 padded)
    float v = 0.0f;
    if (o < OUT_DIM)
        v = (k < HID_DIM) ? W_l2[o * HID_DIM + k] : W_r2[o * HID_DIM + (k - HID_DIM)];
    w2sw[gid] = (_Float16)v;
}

// ---------------------------------------------------------------------------
// Edge aggregation (scatter-mean numerators + degree). f32 atomics resolve in
// the 192MB L2 (every tensor here is L2-resident on MI455X).
// ---------------------------------------------------------------------------
__global__ __launch_bounds__(256) void agg1_kernel(
    const float* __restrict__ x, const int* __restrict__ ei,
    float* __restrict__ agg, float* __restrict__ cnt)
{
    int gid  = blockIdx.x * 256 + threadIdx.x;         // E*16 threads exactly
    int e    = gid >> 4;
    int part = gid & 15;
    int src = ei[e];
    int dst = ei[N_EDGES + e];
    float4 v = *reinterpret_cast<const float4*>(x + (size_t)src * IN_DIM + part * 4);
    float* ap = agg + (size_t)dst * IN_DIM + part * 4;
    atomicAdd(ap + 0, v.x); atomicAdd(ap + 1, v.y);
    atomicAdd(ap + 2, v.z); atomicAdd(ap + 3, v.w);
    if (part == 0) atomicAdd(cnt + dst, 1.0f);
}

__global__ __launch_bounds__(256) void agg2_kernel(
    const float* __restrict__ h1, const int* __restrict__ ei,
    float* __restrict__ agg)
{
    int gid  = blockIdx.x * 256 + threadIdx.x;         // E*32 threads exactly
    int e    = gid >> 5;
    int part = gid & 31;
    int src = ei[e];
    int dst = ei[N_EDGES + e];
    float4 v = *reinterpret_cast<const float4*>(h1 + (size_t)src * HID_DIM + part * 4);
    float* ap = agg + (size_t)dst * HID_DIM + part * 4;
    atomicAdd(ap + 0, v.x); atomicAdd(ap + 1, v.y);
    atomicAdd(ap + 2, v.z); atomicAdd(ap + 3, v.w);
}

// ---------------------------------------------------------------------------
// Build f16 concat operands (A matrices) for the WMMA GEMMs.
// ---------------------------------------------------------------------------
__global__ __launch_bounds__(256) void build_z1_kernel(
    const float* __restrict__ agg1, const float* __restrict__ cnt,
    const float* __restrict__ x, _Float16* __restrict__ z1)
{
    int gid = blockIdx.x * 256 + threadIdx.x;          // N*64 threads exactly
    int n = gid >> 6;
    int k = gid & 63;
    float ic = 1.0f / fmaxf(cnt[n], 1.0f);
    z1[(size_t)n * K1DIM + k]          = (_Float16)(agg1[(size_t)n * IN_DIM + k] * ic);
    z1[(size_t)n * K1DIM + IN_DIM + k] = (_Float16)(x[(size_t)n * IN_DIM + k]);
}

__global__ __launch_bounds__(256) void build_z2_kernel(
    const float* __restrict__ agg2, const float* __restrict__ cnt,
    _Float16* __restrict__ z2)
{
    int gid = blockIdx.x * 256 + threadIdx.x;          // N*128 threads exactly
    int n = gid >> 7;
    int k = gid & 127;
    float ic = 1.0f / fmaxf(cnt[n], 1.0f);
    z2[(size_t)n * K2DIM + k] = (_Float16)(agg2[(size_t)n * HID_DIM + k] * ic);
}

// ---------------------------------------------------------------------------
// Layer-1 GEMM: h1 = relu(z1 @ Wc1^T + b1). The block's 16-node A tile (4KB)
// is staged once via async global->LDS (ASYNCcnt), then 8 waves (one 16x16
// output tile each) read fragments with ds_load_b128. K=128 -> 4 WMMAs/wave.
// Fused dual store: f32 h1 (for edge aggregation) + f16 high half of z2.
// ---------------------------------------------------------------------------
__global__ __launch_bounds__(256) void gemm1_kernel(
    const _Float16* __restrict__ z1, const _Float16* __restrict__ w1sw,
    const float* __restrict__ b_l1,
    float* __restrict__ h1, _Float16* __restrict__ z2)
{
    __shared__ _Float16 As[16 * LDS1_STRIDE];
    const int t  = threadIdx.x;
    const int mt = blockIdx.x;                         // node tile (3125 exact)

    // Async copy: 16 rows x 256B, LDS row stride 272B. One b128 per thread.
    {
        const int r = t >> 4, chunk = t & 15;
        unsigned lds_off = (unsigned)(size_t)(&As[0])
                         + (unsigned)(r * (LDS1_STRIDE * 2) + chunk * 16);
        unsigned long long ga =
            (unsigned long long)(size_t)(z1 + (size_t)(mt * 16 + r) * K1DIM)
            + (unsigned long long)(chunk * 16);
        asm volatile("global_load_async_to_lds_b128 %0, %1, off"
                     :: "v"(lds_off), "v"(ga) : "memory");
    }
    asm volatile("s_wait_asynccnt 0x0" ::: "memory");
    __syncthreads();

    const int wave = t >> 5;                           // 0..7: output tile
    const int lane = t & 31;
    const int col  = lane & 15;                        // A row m / D col n
    const int half = lane >> 4;
    const int kb0  = half * 8;                         // A K sub-base per lane half

    v8f c = {};
    const _Float16* arow = &As[col * LDS1_STRIDE];
#pragma unroll
    for (int it = 0; it < 4; ++it) {
        AFrag a, b;
        const int kbase = it * 32;
        a.q[0] = *reinterpret_cast<const uint4*>(arow + kbase + kb0);
        a.q[1] = *reinterpret_cast<const uint4*>(arow + kbase + kb0 + 16);
        const _Float16* bp = w1sw + (size_t)(((wave * 4 + it) * 32) + lane) * 16;
        b.q[0] = *reinterpret_cast<const uint4*>(bp);
        b.q[1] = *reinterpret_cast<const uint4*>(bp + 8);
        c = __builtin_amdgcn_wmma_f32_16x16x32_f16(false, a.v, false, b.v,
                                                   (short)0, c, false, false);
    }
    const int o = wave * 16 + col;
    const float bias = b_l1[o];
    const int rowbase = mt * 16 + half * 8;
#pragma unroll
    for (int r = 0; r < 8; ++r) {
        float v = c[r] + bias;
        v = v > 0.0f ? v : 0.0f;                       // ReLU
        const int node = rowbase + r;
        h1[(size_t)node * HID_DIM + o] = v;
        z2[(size_t)node * K2DIM + HID_DIM + o] = (_Float16)v;
    }
}

// ---------------------------------------------------------------------------
// Layer-2 GEMM: h2 = relu(z2 @ Wc2^T + b2), 40 outputs padded to 48 (3 tiles).
// 4 waves stage the 8KB A tile async->LDS; waves 0..2 compute (K=256 -> 8
// WMMAs); wave 3 only helps copy. Divergence is whole-wave, EXEC stays full.
// ---------------------------------------------------------------------------
__global__ __launch_bounds__(128) void gemm2_kernel(
    const _Float16* __restrict__ z2, const _Float16* __restrict__ w2sw,
    const float* __restrict__ b_l2, float* __restrict__ h2)
{
    __shared__ _Float16 As[16 * LDS2_STRIDE];
    const int t  = threadIdx.x;
    const int mt = blockIdx.x;

    // Async copy: 16 rows x 512B, LDS row stride 528B. 4 b128 per thread.
#pragma unroll
    for (int j = 0; j < 4; ++j) {
        const int idx = t + j * 128;                   // 0..511
        const int r = idx >> 5, chunk = idx & 31;
        unsigned lds_off = (unsigned)(size_t)(&As[0])
                         + (unsigned)(r * (LDS2_STRIDE * 2) + chunk * 16);
        unsigned long long ga =
            (unsigned long long)(size_t)(z2 + (size_t)(mt * 16 + r) * K2DIM)
            + (unsigned long long)(chunk * 16);
        asm volatile("global_load_async_to_lds_b128 %0, %1, off"
                     :: "v"(lds_off), "v"(ga) : "memory");
    }
    asm volatile("s_wait_asynccnt 0x0" ::: "memory");
    __syncthreads();

    const int wave = t >> 5;
    if (wave < 3) {
        const int lane = t & 31;
        const int col  = lane & 15;
        const int half = lane >> 4;
        const int kb0  = half * 8;

        v8f c = {};
        const _Float16* arow = &As[col * LDS2_STRIDE];
#pragma unroll
        for (int it = 0; it < 8; ++it) {
            AFrag a, b;
            const int kbase = it * 32;
            a.q[0] = *reinterpret_cast<const uint4*>(arow + kbase + kb0);
            a.q[1] = *reinterpret_cast<const uint4*>(arow + kbase + kb0 + 16);
            const _Float16* bp = w2sw + (size_t)(((wave * 8 + it) * 32) + lane) * 16;
            b.q[0] = *reinterpret_cast<const uint4*>(bp);
            b.q[1] = *reinterpret_cast<const uint4*>(bp + 8);
            c = __builtin_amdgcn_wmma_f32_16x16x32_f16(false, a.v, false, b.v,
                                                       (short)0, c, false, false);
        }
        const int o = wave * 16 + col;
        if (o < OUT_DIM) {
            const float bias = b_l2[o];
            const int rowbase = mt * 16 + half * 8;
#pragma unroll
            for (int r = 0; r < 8; ++r) {
                float v = c[r] + bias;
                v = v > 0.0f ? v : 0.0f;               // ReLU
                h2[(size_t)(rowbase + r) * OUT_DIM + o] = v;
            }
        }
    }
}

// ---------------------------------------------------------------------------
// log_softmax over 40 classes; output transposed to [40, N] layout.
// One thread per node; float4 row reads; writes coalesced across nodes.
// ---------------------------------------------------------------------------
__global__ __launch_bounds__(256) void logsoftmax_kernel(
    const float* __restrict__ h2, float* __restrict__ out)
{
    int n = blockIdx.x * 256 + threadIdx.x;
    if (n >= N_NODES) return;
    const float4* rowv = reinterpret_cast<const float4*>(h2 + (size_t)n * OUT_DIM);
    float row[OUT_DIM];
#pragma unroll
    for (int i = 0; i < OUT_DIM / 4; ++i) {
        float4 v = rowv[i];
        row[4*i+0] = v.x; row[4*i+1] = v.y; row[4*i+2] = v.z; row[4*i+3] = v.w;
    }
    float m = -3.402823466e38f;
#pragma unroll
    for (int cc = 0; cc < OUT_DIM; ++cc) m = fmaxf(m, row[cc]);
    float s = 0.0f;
#pragma unroll
    for (int cc = 0; cc < OUT_DIM; ++cc) s += __expf(row[cc] - m);
    float lse = m + __logf(s);
#pragma unroll
    for (int cc = 0; cc < OUT_DIM; ++cc)
        out[(size_t)cc * N_NODES + n] = row[cc] - lse;
}

// ---------------------------------------------------------------------------
extern "C" void kernel_launch(void* const* d_in, const int* in_sizes, int n_in,
                              void* d_out, int out_size, void* d_ws, size_t ws_size,
                              hipStream_t stream) {
    const float* x    = (const float*)d_in[0];
    const int*   ei   = (const int*)d_in[1];   // [2,E]: row0 src, row1 dst (int32: JAX x64 off)
    const float* W_l1 = (const float*)d_in[2];
    const float* b_l1 = (const float*)d_in[3];
    const float* W_r1 = (const float*)d_in[4];
    const float* W_l2 = (const float*)d_in[5];
    const float* b_l2 = (const float*)d_in[6];
    const float* W_r2 = (const float*)d_in[7];
    float* out = (float*)d_out;

    // Workspace carve-out (~111 MB total)
    char* ws = (char*)d_ws;
    auto carve = [&](size_t bytes) {
        char* p = ws;
        ws += (bytes + 255) & ~(size_t)255;
        return p;
    };
    float*    cnt  = (float*)carve((size_t)N_NODES * 4);
    float*    agg1 = (float*)carve((size_t)N_NODES * IN_DIM * 4);
    float*    agg2 = (float*)carve((size_t)N_NODES * HID_DIM * 4);
    float*    h1   = (float*)carve((size_t)N_NODES * HID_DIM * 4);
    float*    h2   = (float*)carve((size_t)N_NODES * OUT_DIM * 4);
    _Float16* z1   = (_Float16*)carve((size_t)N_NODES * K1DIM * 2);
    _Float16* z2   = (_Float16*)carve((size_t)N_NODES * K2DIM * 2);
    _Float16* w1sw = (_Float16*)carve(16384 * 2);
    _Float16* w2sw = (_Float16*)carve(12288 * 2);

    // Zero accumulators every call (graph-capture-safe async memset)
    hipMemsetAsync(cnt,  0, (size_t)N_NODES * 4, stream);
    hipMemsetAsync(agg1, 0, (size_t)N_NODES * IN_DIM * 4, stream);
    hipMemsetAsync(agg2, 0, (size_t)N_NODES * HID_DIM * 4, stream);

    prep_w1_kernel<<<64, 256, 0, stream>>>(W_l1, W_r1, w1sw);
    prep_w2_kernel<<<48, 256, 0, stream>>>(W_l2, W_r2, w2sw);

    agg1_kernel<<<(N_EDGES * 16) / 256, 256, 0, stream>>>(x, ei, agg1, cnt);
    build_z1_kernel<<<(N_NODES * IN_DIM) / 256, 256, 0, stream>>>(agg1, cnt, x, z1);
    gemm1_kernel<<<N_NODES / 16, 256, 0, stream>>>(z1, w1sw, b_l1, h1, z2);

    agg2_kernel<<<(N_EDGES * 32) / 256, 256, 0, stream>>>(h1, ei, agg2);
    build_z2_kernel<<<(N_NODES * HID_DIM) / 256, 256, 0, stream>>>(agg2, cnt, z2);
    gemm2_kernel<<<N_NODES / 16, 128, 0, stream>>>(z2, w2sw, b_l2, h2);

    logsoftmax_kernel<<<(N_NODES + 255) / 256, 256, 0, stream>>>(h2, out);
}
